// CQCCExtractor_23948737642977
// MI455X (gfx1250) — matrix operand order repaired and checked by reading, not compiled, not run on hardware
//
#include <hip/hip_runtime.h>
#include <hip/hip_bf16.h>
#include <math.h>

typedef __attribute__((ext_vector_type(16))) _Float16 v16h;
typedef __attribute__((ext_vector_type(8)))  float    v8f;

#define NW     262144      // wav length
#define PADL   32768       // N_FFT/2 reflect pad each side
#define HOP    128
#define TFRM   2049        // 1 + NW/HOP
#define MTILES 129         // ceil(TFRM/16)
#define MT     (MTILES*16) // 2064
#define NBINS  96
#define NCOL   384         // 96 bins * (lo,hi) * (re,im)
#define NTILES 24          // NCOL/16
#define KP     16768       // 131*128 >= MIN_WIN
#define KB     131         // KP/128
#define KCH    524         // KP/32
#define NFFT   65536
#define TWO_PI 6.283185307179586
// xh holds x_pad[left + i]: frame t, tap k -> xh[t*HOP + k]; max index 128*2048+15*128+16767
#define XH2    280832
#define LSTR   136         // LDS row stride in halves: 272B rows, 16B aligned

#if __has_builtin(__builtin_amdgcn_global_load_async_to_lds_b128)
#define HAVE_ASYNC_LDS 1
// exact builtin param types: (v4i32 as1*, v4i32 as3*, imm int, imm int)
typedef int v4i_vs __attribute__((vector_size(16)));
typedef __attribute__((address_space(1))) v4i_vs* gasync_p;
typedef __attribute__((address_space(3))) v4i_vs* lasync_p;
#else
#define HAVE_ASYNC_LDS 0
#endif

__device__ __forceinline__ void wait_async_lds() {
#if __has_builtin(__builtin_amdgcn_s_wait_asynccnt)
    __builtin_amdgcn_s_wait_asynccnt(0);
#else
    asm volatile("s_wait_asynccnt 0x0" ::: "memory");
#endif
}

// ---------------- 1) reflect pad + cast to f16, with `left` folded in ----------------
__global__ void k_pad(const float* __restrict__ wav, _Float16* __restrict__ xh, int left) {
    int i = blockIdx.x * 256 + threadIdx.x;
    if (i >= XH2) return;
    int s = i + left - PADL;
    if (s < 0) s = -s;
    if (s >= NW) s = 2 * NW - 2 - s;
    xh[i] = (_Float16)wav[s];
}

// ---------------- 2) basis in WMMA B-fragment layout ----------------
// bf[((kch*24 + nt)*32 + lane)*16 + 2v + p] = B[kch*32 + (lane>>4)*16 + 2v + p][nt*16 + (lane&15)]
__global__ void k_basis(_Float16* __restrict__ bf, int min_win, int left) {
    const int TOT = KCH * NTILES * 32 * 8;
    int i = blockIdx.x * 256 + threadIdx.x;
    if (i >= TOT) return;
    int v    = i & 7;
    int lane = (i >> 3) & 31;
    int rem  = i >> 8;
    int nt   = rem % NTILES;
    int kch  = rem / NTILES;
    int col  = nt * 16 + (lane & 15);
    int b    = col >> 2, which = col & 3;
    double cqt  = 32.7 * exp2((double)b * (1.0 / 24.0));
    double idxf = cqt * (32768.0 / 8000.0);           // /(SR/2)*(NUM_FREQ-1)
    int lo = (int)idxf; lo = lo < 0 ? 0 : (lo > 32767 ? 32767 : lo);
    int fi = (which < 2) ? lo : lo + 1;
    int comp = which & 1;                              // 0=cos(Re), 1=sin(|Im|)
    int kk0 = kch * 32 + (lane >> 4) * 16 + 2 * v;
    int base = (((kch * NTILES + nt) * 32 + lane) * 16) + 2 * v;
#pragma unroll
    for (int p = 0; p < 2; ++p) {
        int kk = kk0 + p;
        double val = 0.0;
        if (kk < min_win) {
            double w = 0.5 - 0.5 * cos(TWO_PI * (double)kk / (double)min_win);
            long long ph = ((long long)fi * (long long)(left + kk)) & (long long)(NFFT - 1);
            double th = (TWO_PI / (double)NFFT) * (double)ph;  // exact phase mod 2*pi
            val = w * (comp ? sin(th) : cos(th));
        }
        bf[base + p] = (_Float16)val;
    }
}

// ---------------- 3) WMMA GEMM: [MT x KP] * [KP x NCOL] -> gout f32 ----------------
__global__ __launch_bounds__(256) void k_gemm(const _Float16* __restrict__ xh,
                                              const _Float16* __restrict__ bf,
                                              float* __restrict__ gout) {
    __shared__ _Float16 lds_a[2][16 * LSTR];   // double-buffered A tile, 8.7 KB
    const int tid   = threadIdx.x;
    const int lane  = tid & 31;
    const int wid   = tid >> 5;                    // 0..7
    const int ntile = blockIdx.y * 8 + wid;        // 0..23
    const int mtile = blockIdx.x;                  // 0..128
    const int m     = lane & 15;
    const int sel   = lane >> 4;
    v8f acc = {};

    // each thread stages one 16-byte chunk of the 16x128 A tile per K-block
    const int sr = tid >> 4;            // row 0..15
    const int sc = (tid & 15) * 8;      // col 0..120 (halves)
    const _Float16* gsrc0 = xh + mtile * 16 * HOP + sr * HOP + sc;  // 16B aligned
    _Float16* lrow0 = &lds_a[0][sr * LSTR + sc];                    // 16B aligned
    _Float16* lrow1 = &lds_a[1][sr * LSTR + sc];

    // prologue: stage block 0 into buffer 0
#if HAVE_ASYNC_LDS
    __builtin_amdgcn_global_load_async_to_lds_b128((gasync_p)gsrc0, (lasync_p)lrow0, 0, 0);
    wait_async_lds();
#else
    *(uint4*)lrow0 = *(const uint4*)gsrc0;
#endif
    __syncthreads();

    int buf = 0;
    for (int kb = 0; kb < KB; ++kb) {
        const bool more = (kb + 1 < KB);
#if HAVE_ASYNC_LDS
        if (more) {   // async-copy next tile into the other buffer (ASYNCcnt)
            __builtin_amdgcn_global_load_async_to_lds_b128(
                (gasync_p)(gsrc0 + (kb + 1) * 128),
                (lasync_p)(buf ? lrow0 : lrow1), 0, 0);
        }
#else
        uint4 nxt;
        if (more) nxt = *(const uint4*)(gsrc0 + (kb + 1) * 128);
#endif
        if (more) {   // prefetch next B fragments into L2 (global_prefetch_b8)
            __builtin_prefetch(bf + (size_t)((((kb + 1) * 4) * NTILES + ntile) * 32 + lane) * 16, 0, 1);
        }
#pragma unroll
        for (int kc = 0; kc < 4; ++kc) {
            union { v16h v; _Float16 h[16]; } a;
            const _Float16* rp = &lds_a[buf][m * LSTR + kc * 32 + sel * 16];
#pragma unroll
            for (int j = 0; j < 8; ++j) a.h[j] = rp[j];           // K = sel*16 + 0..7
#pragma unroll
            for (int j = 0; j < 8; ++j) a.h[8 + j] = rp[8 + j];   // K = sel*16 + 8..15
            const v16h b = *(const v16h*)(bf + (size_t)(((kb * 4 + kc) * NTILES + ntile) * 32 + lane) * 16);
            acc = __builtin_amdgcn_wmma_f32_16x16x32_f16(false, a.v, false, b, (short)0, acc, false, false);
        }
#if HAVE_ASYNC_LDS
        wait_async_lds();
#else
        if (more) *(uint4*)(buf ? lrow0 : lrow1) = nxt;
#endif
        __syncthreads();
        buf ^= 1;
    }

    const int col = ntile * 16 + m;
#pragma unroll
    for (int v = 0; v < 8; ++v) {
        int row = mtile * 16 + sel * 8 + v;
        gout[(size_t)row * NCOL + col] = acc[v];
    }
}

// ---------------- 4) power -> interp -> log -> DCT ----------------
__global__ __launch_bounds__(64) void k_cqt(const float* __restrict__ gout,
                                            float* __restrict__ craw) {
    __shared__ float dct[20 * 96];
    __shared__ float logv[64][97];
    int tid = threadIdx.x;
    for (int i = tid; i < 20 * 96; i += 64) {
        int k = i / 96, n = i % 96;
        dct[i] = (float)cos(M_PI * (double)k * (double)(2 * n + 1) / 192.0);
    }
    int t = blockIdx.x * 64 + tid;
    if (t < TFRM) {
        for (int b = 0; b < NBINS; ++b) {
            double cqt  = 32.7 * exp2((double)b * (1.0 / 24.0));
            double idxf = cqt * (32768.0 / 8000.0);
            int lo = (int)idxf; lo = lo < 0 ? 0 : (lo > 32767 ? 32767 : lo);
            float alpha = (float)(idxf - (double)lo);
            const float* g = gout + (size_t)t * NCOL + b * 4;
            float plo = g[0] * g[0] + g[1] * g[1];
            float phi = g[2] * g[2] + g[3] * g[3];
            float p = (1.f - alpha) * plo + alpha * phi;
            logv[tid][b] = logf(fmaxf(p, 1e-8f));
        }
    }
    __syncthreads();
    if (t < TFRM) {
        for (int k = 0; k < 20; ++k) {
            float s = 0.f;
            for (int b = 0; b < NBINS; ++b) s += logv[tid][b] * dct[k * 96 + b];
            craw[t * 20 + k] = s;
        }
    }
}

// ---------------- 5) per-coefficient mean / std (ddof=1) ----------------
__global__ __launch_bounds__(256) void k_stats(const float* __restrict__ craw,
                                               float* __restrict__ stats) {
    int k = blockIdx.x;
    __shared__ float ssum[256], ssq[256];
    float s = 0.f, sq = 0.f;
    for (int t = threadIdx.x; t < TFRM; t += 256) {
        float v = craw[t * 20 + k]; s += v; sq += v * v;
    }
    ssum[threadIdx.x] = s; ssq[threadIdx.x] = sq;
    __syncthreads();
    for (int o = 128; o > 0; o >>= 1) {
        if (threadIdx.x < o) {
            ssum[threadIdx.x] += ssum[threadIdx.x + o];
            ssq[threadIdx.x]  += ssq[threadIdx.x + o];
        }
        __syncthreads();
    }
    if (threadIdx.x == 0) {
        float mean = ssum[0] / (float)TFRM;
        float var  = (ssq[0] - (float)TFRM * mean * mean) / (float)(TFRM - 1);
        var = fmaxf(var, 0.f);
        stats[k]      = mean;
        stats[20 + k] = sqrtf(var) + 1e-8f;
    }
}

// ---------------- 6) normalize -> out[:,0:20] ----------------
__global__ void k_norm(const float* __restrict__ craw, const float* __restrict__ stats,
                       float* __restrict__ cnorm, float* __restrict__ out) {
    int i = blockIdx.x * 256 + threadIdx.x;
    if (i >= TFRM * 20) return;
    int t = i / 20, k = i % 20;
    float v = (craw[i] - stats[k]) / stats[20 + k];
    cnorm[i] = v;
    out[t * 60 + k] = v;
}

// ---------------- 7) delta pass (used twice) ----------------
__global__ void k_delta(const float* __restrict__ src, float* __restrict__ dbuf,
                        float* __restrict__ out, int colOff) {
    int i = blockIdx.x * 256 + threadIdx.x;
    if (i >= TFRM * 20) return;
    int t = i / 20, k = i % 20;
    float acc = 0.f;
#pragma unroll
    for (int n = 1; n <= 2; ++n) {
        int tp = t + n; if (tp > TFRM - 1) tp = TFRM - 1;
        int tm = t - n; if (tm < 0) tm = 0;
        acc += (float)n * (src[tp * 20 + k] - src[tm * 20 + k]);
    }
    float v = acc / 10.0f;
    dbuf[i] = v;
    out[t * 60 + colOff + k] = v;
}

extern "C" void kernel_launch(void* const* d_in, const int* in_sizes, int n_in,
                              void* d_out, int out_size, void* d_ws, size_t ws_size,
                              hipStream_t stream) {
    const float* wav = (const float*)d_in[0];
    float* out = (float*)d_out;

    // replicate Python constants exactly
    double q = 1.0 / (pow(2.0, 1.0 / 24.0) - 1.0);
    int min_win = (int)(q * 16000.0 / 32.7);
    if (min_win < 16) min_win = 16;
    int left = (NFFT - min_win) / 2;

    // workspace carve-up (256B aligned)
    char* ws = (char*)d_ws;
    size_t off = 0;
    auto carve = [&](size_t bytes) { char* p = ws + off; off = (off + bytes + 255) & ~(size_t)255; return p; };
    _Float16* xh    = (_Float16*)carve((size_t)XH2 * 2);
    _Float16* bf    = (_Float16*)carve((size_t)KCH * NTILES * 32 * 16 * 2);   // 12.9 MB
    float*    gout  = (float*)   carve((size_t)MT * NCOL * 4);                // 3.2 MB
    float*    craw  = (float*)   carve((size_t)TFRM * 20 * 4);
    float*    stats = (float*)   carve(40 * 4);
    float*    cnorm = (float*)   carve((size_t)TFRM * 20 * 4);
    float*    d1buf = (float*)   carve((size_t)TFRM * 20 * 4);
    float*    d2buf = (float*)   carve((size_t)TFRM * 20 * 4);
    (void)ws_size; (void)in_sizes; (void)n_in; (void)out_size;

    k_pad<<<(XH2 + 255) / 256, 256, 0, stream>>>(wav, xh, left);

    const int TOT = KCH * NTILES * 32 * 8;
    k_basis<<<(TOT + 255) / 256, 256, 0, stream>>>(bf, min_win, left);

    k_gemm<<<dim3(MTILES, 3), 256, 0, stream>>>(xh, bf, gout);

    k_cqt<<<(TFRM + 63) / 64, 64, 0, stream>>>(gout, craw);
    k_stats<<<20, 256, 0, stream>>>(craw, stats);
    k_norm<<<(TFRM * 20 + 255) / 256, 256, 0, stream>>>(craw, stats, cnorm, out);
    k_delta<<<(TFRM * 20 + 255) / 256, 256, 0, stream>>>(cnorm, d1buf, out, 20);
    k_delta<<<(TFRM * 20 + 255) / 256, 256, 0, stream>>>(d1buf, d2buf, out, 40);
}
